// ModuleNet_63161789055603
// MI455X (gfx1250) — matrix operand order repaired and checked by reading, not compiled, not run on hardware
//
#include <hip/hip_runtime.h>

// ---------------- types ----------------
typedef __bf16 bf16_t;
typedef bf16_t v16bf __attribute__((ext_vector_type(16)));
typedef bf16_t v8bf  __attribute__((ext_vector_type(8)));
typedef float  v8f   __attribute__((ext_vector_type(8)));

// ---------------- problem constants ----------------
#define EMB   1024
#define NMOD  8
#define NAUTH 100000
#define NNODE 50000
#define COUT  8
#define BP    4096      // paths
#define KSTEP 4         // intermediate steps (5 GEMM steps total)
#define NSTEP 5
#define UPDW  0.1f

#define ROWS  64        // rows per GEMM block (4 x 16-row WMMA tiles)
// max 64-row tiles per step across all modules (ceil-sum bound)
#define MAX_TILES4 ((BP + NMOD * 63) / 64 + 1)   // 72ish

// ---------------- kernels ----------------

// Convert W [M,E,E] f32 -> bf16
__global__ __launch_bounds__(256) void cvt_w_kernel(const float* __restrict__ W,
                                                    bf16_t* __restrict__ Wbf, int n4) {
    int i = blockIdx.x * 256 + threadIdx.x;
    if (i >= n4) return;
    float4 f = ((const float4*)W)[i];
    bf16_t* d = Wbf + (size_t)i * 4;
    d[0] = (bf16_t)f.x; d[1] = (bf16_t)f.y; d[2] = (bf16_t)f.z; d[3] = (bf16_t)f.w;
}

// x0[b,:] = bf16(author_table[e0[b],:]) ; one thread per 4 elements
__global__ __launch_bounds__(256) void gather_x0_kernel(const float* __restrict__ at,
                                                        const int* __restrict__ e0,
                                                        bf16_t* __restrict__ x0) {
    int tid = blockIdx.x * 256 + threadIdx.x;         // BP*256 threads
    int b  = tid >> 8;
    int c4 = (tid & 255) << 2;
    float4 f = *(const float4*)&at[(size_t)e0[b] * EMB + c4];
    bf16_t* d = x0 + (size_t)b * EMB + c4;
    d[0] = (bf16_t)f.x; d[1] = (bf16_t)f.y; d[2] = (bf16_t)f.z; d[3] = (bf16_t)f.w;
}

// Bucket path rows by module id for every step.
__global__ __launch_bounds__(256) void bucket_kernel(const int* __restrict__ mod_ids,
                                                     int* __restrict__ counts,
                                                     int* __restrict__ order) {
    int b = blockIdx.x * 256 + threadIdx.x;
    if (b >= BP) return;
    for (int s = 0; s < NSTEP; ++s) {
        int m = mod_ids[b * NSTEP + s];
        int p = atomicAdd(&counts[s * NMOD + m], 1);
        order[(size_t)(s * NMOD + m) * BP + p] = b;
    }
}

// Grouped GEMM step: block = 4 waves = 64 rows x 64 cols of one module bucket.
// Each wave owns one 16-col tile and 4 row tiles (4 accumulators share each B fragment):
// 4x less W traffic from L2 and 4 independent WMMAs per B load.
#define A_STRIDE (EMB + 8)   // pad: conflict-free ds_load_b128 across 16 rows
__global__ __launch_bounds__(128)
void path_step_gemm(const bf16_t* __restrict__ Wbf,
                    const bf16_t* __restrict__ xin,
                    const float*  __restrict__ biasTab,   // node_table or author_table
                    const int*    __restrict__ auxIdx,    // node_ids or e_last
                    int auxStride, int auxOff,
                    const int*    __restrict__ counts,    // [NMOD] for this step
                    const int*    __restrict__ order,     // [NMOD][BP]
                    bf16_t* __restrict__ ybf,             // bf16 out (steps 0..3)
                    float*  __restrict__ yf,              // f32 out (final step)
                    int lastStep)
{
    __shared__ bf16_t As[ROWS * A_STRIDE];   // ~132 KB of the 320 KB WGP LDS
    __shared__ int rowB[ROWS];
    __shared__ int aux[ROWS];

    // map blockIdx.x -> (module m, 64-row tile rt)
    int slot = blockIdx.x;
    int m = -1, rt = 0, cnt = 0;
    {
        int acc = 0;
        for (int mm = 0; mm < NMOD; ++mm) {
            int c = counts[mm];
            int t = (c + 63) >> 6;
            if (m < 0 && slot < acc + t) { m = mm; rt = slot - acc; cnt = c; }
            acc += t;
        }
        if (m < 0) return;   // uniform exit; WMMA waves keep EXEC all-ones
    }

    int tid = threadIdx.x;
    if (tid < ROWS) {
        int gr = rt * ROWS + tid;
        int b = (gr < cnt) ? order[(size_t)m * BP + gr] : -1;
        rowB[tid] = b;
        aux[tid] = (b >= 0) ? auxIdx[b * auxStride + auxOff] : 0;
    }
    __syncthreads();

    // A-strip staging: 64 rows x 1024 bf16 via async gather straight into LDS
    // (GLOBAL_LOAD_ASYNC_TO_LDS_B128, tracked by ASYNCcnt; no VGPR round-trip).
    for (int v = tid; v < ROWS * (EMB / 8); v += 128) {
        int r  = v >> 7;
        int kv = (v & 127) << 3;
        int b = rowB[r];
        if (b >= 0) {
            unsigned lds_off = (unsigned)(uintptr_t)&As[r * A_STRIDE + kv];
            unsigned long long ga =
                (unsigned long long)(uintptr_t)&xin[(size_t)b * EMB + kv];
            asm volatile("global_load_async_to_lds_b128 %0, %1, off"
                         :: "v"(lds_off), "v"(ga) : "memory");
        }
    }
    asm volatile("s_wait_asynccnt 0x0" ::: "memory");
    __syncthreads();

    int wave = tid >> 5;
    int lane = tid & 31;
    int n    = lane & 15;
    int half = lane >> 4;
    int o0   = blockIdx.y * 64 + wave * 16;

    // ISA 7.12.2 layouts (bf16, wave32):
    //  A 16x32: lane(0-15)=row, elems 0..7 -> K k0+8h..+7, elems 8..15 -> K k0+16+8h..
    //  B 32x16: lane(0-15)=col, elems 0..7 -> K k0+8h..+7, elems 8..15 -> K k0+16+8h.. (same as A)
    const bf16_t* Abase = &As[n * A_STRIDE + half * 8];
    const bf16_t* Bbase = Wbf + ((size_t)m << 20) + (size_t)(o0 + n) * EMB + half * 8;

    v8f z = {0.f, 0.f, 0.f, 0.f, 0.f, 0.f, 0.f, 0.f};
    v8f acc[4];
    acc[0] = z; acc[1] = z; acc[2] = z; acc[3] = z;

    #pragma unroll 2
    for (int k0 = 0; k0 < EMB; k0 += 32) {
        union { v16bf v; v8bf h[2]; } bm;
        bm.h[0] = *(const v8bf*)(Bbase + k0);
        bm.h[1] = *(const v8bf*)(Bbase + k0 + 16);
        #pragma unroll
        for (int rt4 = 0; rt4 < 4; ++rt4) {
            union { v16bf v; v8bf h[2]; } a;
            const bf16_t* Ab = Abase + rt4 * 16 * A_STRIDE + k0;
            a.h[0] = *(const v8bf*)(Ab);
            a.h[1] = *(const v8bf*)(Ab + 16);
            acc[rt4] = __builtin_amdgcn_wmma_f32_16x16x32_bf16(
                           false, a.v, false, bm.v, (short)0, acc[rt4], false, false);
        }
    }

    // D layout: VGPR r -> M=r (lanes 0-15) / M=8+r (lanes 16-31), N=lane&15
    int col = o0 + n;
    #pragma unroll
    for (int rt4 = 0; rt4 < 4; ++rt4) {
        #pragma unroll
        for (int r = 0; r < 8; ++r) {
            int Mrow = rt4 * 16 + half * 8 + r;
            int b = rowB[Mrow];
            if (b < 0) continue;
            float v = acc[rt4][r];
            float bias = biasTab[(size_t)aux[Mrow] * EMB + col];
            if (!lastStep) {
                v += bias;
                v = v > 0.f ? v : 0.f;
                ybf[(size_t)b * EMB + col] = (bf16_t)v;
            } else {
                yf[(size_t)b * EMB + col] = v + bias;
            }
        }
    }
}

// winner[e] = max b with e_last[b]==e  (duplicate scatter: last index wins, like JAX)
__global__ __launch_bounds__(256) void win_kernel(const int* __restrict__ e_last,
                                                  int* __restrict__ win) {
    int b = blockIdx.x * 256 + threadIdx.x;
    if (b < BP) atomicMax(&win[e_last[b]], b);
}

// tableOut[e_last[b],:] = 0.9*old + 0.1*xf[b,:]  (only winning b writes)
__global__ __launch_bounds__(256) void update_kernel(const int* __restrict__ e_last,
                                                     const int* __restrict__ win,
                                                     const float* __restrict__ at,
                                                     const float* __restrict__ xf,
                                                     float* __restrict__ tableOut) {
    int b = blockIdx.x;
    int row = e_last[b];
    if (win[row] != b) return;
    for (int col = threadIdx.x; col < EMB; col += 256) {
        float o  = at[(size_t)row * EMB + col];
        float nx = xf[(size_t)b * EMB + col];
        tableOut[(size_t)row * EMB + col] = (1.0f - UPDW) * o + UPDW * nx;
    }
}

// logits[b,:] = xf[b,:] @ Wc^T + bc
__global__ __launch_bounds__(256) void logits_kernel(const float* __restrict__ xf,
                                                     const float* __restrict__ Wc,
                                                     const float* __restrict__ bc,
                                                     float* __restrict__ out) {
    __shared__ float red[256 * COUT];
    int b = blockIdx.x, t = threadIdx.x;
    float p[COUT];
    #pragma unroll
    for (int c = 0; c < COUT; ++c) p[c] = 0.f;
    for (int col = t; col < EMB; col += 256) {
        float xv = xf[(size_t)b * EMB + col];
        #pragma unroll
        for (int c = 0; c < COUT; ++c) p[c] += xv * Wc[c * EMB + col];
    }
    #pragma unroll
    for (int c = 0; c < COUT; ++c) red[t * COUT + c] = p[c];
    __syncthreads();
    if (t < COUT) {
        float s = bc[t];
        for (int i = 0; i < 256; ++i) s += red[i * COUT + t];
        out[b * COUT + t] = s;
    }
}

// ---------------- launch ----------------
extern "C" void kernel_launch(void* const* d_in, const int* in_sizes, int n_in,
                              void* d_out, int out_size, void* d_ws, size_t ws_size,
                              hipStream_t stream) {
    const float* author_table = (const float*)d_in[0];
    const float* node_table   = (const float*)d_in[1];
    const float* W            = (const float*)d_in[2];
    const float* Wc           = (const float*)d_in[3];
    const float* bc           = (const float*)d_in[4];
    const int*   e0           = (const int*)d_in[5];
    const int*   mod_ids      = (const int*)d_in[6];
    const int*   node_ids     = (const int*)d_in[7];
    const int*   e_last       = (const int*)d_in[8];

    float* logits_out = (float*)d_out;                       // [BP, COUT]
    float* table_out  = (float*)d_out + (size_t)BP * COUT;   // [NAUTH, EMB]

    // workspace layout
    char* ws = (char*)d_ws;
    bf16_t* Wbf   = (bf16_t*)(ws);                                        // 16 MB
    bf16_t* x0    = (bf16_t*)(ws + (size_t)16 * 1024 * 1024);             // 8 MB
    bf16_t* x1    = (bf16_t*)(ws + (size_t)24 * 1024 * 1024);             // 8 MB
    float*  xf    = (float*) (ws + (size_t)32 * 1024 * 1024);             // 16 MB
    int*    counts= (int*)   (ws + (size_t)48 * 1024 * 1024);             // 5*8 ints
    int*    order = (int*)   (ws + (size_t)48 * 1024 * 1024 + 4096);      // 5*8*BP ints
    int*    win   = (int*)   (ws + (size_t)48 * 1024 * 1024 + 4096
                                 + (size_t)NSTEP * NMOD * BP * 4);        // NAUTH ints

    // init
    hipMemsetAsync(counts, 0, NSTEP * NMOD * sizeof(int), stream);
    hipMemsetAsync(win, 0xFF, (size_t)NAUTH * sizeof(int), stream);

    // bulk copy of author table into output (scatter rows overwritten later)
    hipMemcpyAsync(table_out, author_table, (size_t)NAUTH * EMB * sizeof(float),
                   hipMemcpyDeviceToDevice, stream);

    // precision prep + bucketing
    {
        int n4 = NMOD * EMB * EMB / 4;
        cvt_w_kernel<<<(n4 + 255) / 256, 256, 0, stream>>>(W, Wbf, n4);
        gather_x0_kernel<<<BP, 256, 0, stream>>>(author_table, e0, x0);
        bucket_kernel<<<(BP + 255) / 256, 256, 0, stream>>>(mod_ids, counts, order);
    }

    // 5 grouped-GEMM steps, ping-pong x0/x1, final writes f32 xf
    dim3 grid(MAX_TILES4, EMB / 64);
    for (int s = 0; s < NSTEP; ++s) {
        const bf16_t* xin = (s & 1) ? x1 : x0;
        bf16_t* yout      = (s & 1) ? x0 : x1;
        int last = (s == NSTEP - 1);
        const float* biasTab = last ? author_table : node_table;
        const int*   auxIdx  = last ? e_last : node_ids;
        int auxStride        = last ? 1 : KSTEP;
        int auxOff           = last ? 0 : s;
        path_step_gemm<<<grid, 128, 0, stream>>>(
            Wbf, xin, biasTab, auxIdx, auxStride, auxOff,
            counts + s * NMOD, order + (size_t)s * NMOD * BP,
            yout, xf, last);
    }

    // deterministic duplicate-safe scatter update
    win_kernel<<<(BP + 255) / 256, 256, 0, stream>>>(e_last, win);
    update_kernel<<<BP, 256, 0, stream>>>(e_last, win, author_table, xf, table_out);

    // classifier head
    logits_kernel<<<BP, 256, 0, stream>>>(xf, Wc, bc, logits_out);
}
